// RotaryTransformerBlock_1125281432028
// MI455X (gfx1250) — matrix operand order repaired and checked by reading, not compiled, not run on hardware
//
#include <hip/hip_runtime.h>
#include <hip/hip_bf16.h>
#include <math.h>

typedef __bf16 bf16;
typedef __attribute__((ext_vector_type(16))) __bf16 v16bf;
typedef __attribute__((ext_vector_type(8)))  float  v8f;
typedef __attribute__((ext_vector_type(4)))  int    v4i;

union Frag {
    uint4 u[2];
    v16bf v;
};

// ---- gfx1250 async global->LDS path (probe-verified this toolchain) ----
#define TRY_ASYNC 1
#if TRY_ASYNC && defined(__has_builtin)
#  if __has_builtin(__builtin_amdgcn_global_load_async_to_lds_b128)
#    define ASYNC_OK 1
#  endif
#endif
#ifndef ASYNC_OK
#  define ASYNC_OK 0
#endif

#if ASYNC_OK
typedef __attribute__((address_space(1))) v4i Gv4i;  // global 16B chunk
typedef __attribute__((address_space(3))) v4i Lv4i;  // LDS 16B chunk
#endif

__device__ __forceinline__ void copy16(const bf16* __restrict__ g, bf16* __restrict__ l)
{
#if ASYNC_OK
    __builtin_amdgcn_global_load_async_to_lds_b128((Gv4i*)g, (Lv4i*)l, 0, 0);
#else
    *(uint4*)l = *(const uint4*)g;
#endif
}

__device__ __forceinline__ void async_fence()
{
#if ASYNC_OK
    asm volatile("s_wait_asynccnt 0x0" ::: "memory");
#endif
}

// ---------------------------------------------------------------------------
// Weight prep: W f32[K,N] -> Wt bf16[N,K]  (one-time per launch)
// ---------------------------------------------------------------------------
__global__ __launch_bounds__(256) void wprep_kernel(
    const float* __restrict__ W, bf16* __restrict__ Wt, int K, int N)
{
    __shared__ float tile[32][33];
    const int n0 = blockIdx.x * 32, k0 = blockIdx.y * 32;
    const int tx = threadIdx.x & 31, ty = threadIdx.x >> 5; // 32 x 8
#pragma unroll
    for (int j = 0; j < 4; j++) {
        int k = ty + j * 8;
        tile[k][tx] = W[(size_t)(k0 + k) * N + n0 + tx];
    }
    __syncthreads();
#pragma unroll
    for (int j = 0; j < 4; j++) {
        int n = ty + j * 8;
        Wt[(size_t)(n0 + n) * K + k0 + tx] = (bf16)tile[tx][n];
    }
}

// ---------------------------------------------------------------------------
// bf16 WMMA GEMM, double-buffered async tiles.
// BM=BN=128, BK=64, 256 thr = 8 waves (4m x 2n), wave tile 32x64.
// MODE 0: +bias -> bf16 | MODE 1: +bias+resid -> f32 | MODE 2: +bias,GELU -> bf16
// ---------------------------------------------------------------------------
constexpr int BM = 128, BN = 128, BK = 64, PK = 72; // 144B pitch, 16B aligned

template <int MODE>
__global__ __launch_bounds__(256) void gemm_bf16(
    const bf16* __restrict__ A, const bf16* __restrict__ Wt,
    const float* __restrict__ bias, const float* __restrict__ resid,
    void* __restrict__ out, int M, int N, int K)
{
    __shared__ bf16 As[2][BM * PK];
    __shared__ bf16 Bs[2][BN * PK];

    const int tid = threadIdx.x;
    const int l   = tid & 31;
    const int w   = tid >> 5;
    const int m0  = blockIdx.y * BM;
    const int n0  = blockIdx.x * BN;
    const int wm  = (w >> 1) * 32;
    const int wn  = (w & 1)  * 64;

    const int fr = l & 15;
    const int kc = (l >> 4) << 3;    // A-frag chunk base (0/8)
    const int kb = (l >> 4) << 4;    // unused (0/16)
    const int rh = (l >> 4) << 3;    // D-frag row-half

    auto load_tiles = [&](int k0, int buf) {
#pragma unroll
        for (int p = 0; p < 4; p++) {
            int c = p * 256 + tid;
            int row = c >> 3, col = (c & 7) * 8;
            copy16(&A[(size_t)(m0 + row) * K + k0 + col], &As[buf][row * PK + col]);
        }
#pragma unroll
        for (int p = 0; p < 4; p++) {
            int c = p * 256 + tid;
            int row = c >> 3, col = (c & 7) * 8;
            copy16(&Wt[(size_t)(n0 + row) * K + k0 + col], &Bs[buf][row * PK + col]);
        }
    };

    v8f acc[2][4];
#pragma unroll
    for (int i = 0; i < 2; i++)
#pragma unroll
        for (int j = 0; j < 4; j++) acc[i][j] = {};

    const int nk = K / BK;
    load_tiles(0, 0);

    for (int kt = 0; kt < nk; kt++) {
        async_fence();
        __syncthreads();                     // tile kt published; buf kt-1 free
        if (kt + 1 < nk) load_tiles((kt + 1) * BK, (kt + 1) & 1);
        const int buf = kt & 1;

#pragma unroll
        for (int ks = 0; ks < 2; ks++) {
            Frag a[2], b[4];
#pragma unroll
            for (int i = 0; i < 2; i++) {
                const bf16* base = &As[buf][(wm + i * 16 + fr) * PK + ks * 32 + kc];
                a[i].u[0] = *(const uint4*)base;
                a[i].u[1] = *(const uint4*)(base + 16);
            }
#pragma unroll
            for (int j = 0; j < 4; j++) {
                const bf16* base = &Bs[buf][(wn + j * 16 + fr) * PK + ks * 32 + kc];
                b[j].u[0] = *(const uint4*)base;
                b[j].u[1] = *(const uint4*)(base + 16);
            }
#pragma unroll
            for (int i = 0; i < 2; i++)
#pragma unroll
                for (int j = 0; j < 4; j++)
                    acc[i][j] = __builtin_amdgcn_wmma_f32_16x16x32_bf16(
                        false, a[i].v, false, b[j].v, (short)0, acc[i][j], false, false);
        }
    }

#pragma unroll
    for (int i = 0; i < 2; i++)
#pragma unroll
        for (int j = 0; j < 4; j++)
#pragma unroll
            for (int r = 0; r < 8; r++) {
                int row = m0 + wm + i * 16 + rh + r;
                int col = n0 + wn + j * 16 + fr;
                float v = acc[i][j][r] + bias[col];
                size_t idx = (size_t)row * N + col;
                if constexpr (MODE == 0) {
                    ((bf16*)out)[idx] = (bf16)v;
                } else if constexpr (MODE == 1) {
                    ((float*)out)[idx] = v + resid[idx];
                } else {
                    float g = 0.5f * v * (1.0f + erff(v * 0.70710678118654752f));
                    ((bf16*)out)[idx] = (bf16)g;
                }
            }
}

// ---------------------------------------------------------------------------
// LayerNorm (row length L), fp32 or bf16 input -> bf16 output
// ---------------------------------------------------------------------------
template <typename Tin>
__global__ __launch_bounds__(256) void ln_kernel(
    const Tin* __restrict__ x, const float* __restrict__ g,
    const float* __restrict__ bb, bf16* __restrict__ out, int L)
{
    __shared__ float redA[256];
    __shared__ float redB[256];
    const size_t row = blockIdx.x;
    const Tin* xr = x + row * (size_t)L;
    float s = 0.f, s2 = 0.f;
    for (int i = threadIdx.x; i < L; i += 256) {
        float v = (float)xr[i];
        s += v; s2 += v * v;
    }
    redA[threadIdx.x] = s;
    redB[threadIdx.x] = s2;
    __syncthreads();
    for (int st = 128; st > 0; st >>= 1) {
        if (threadIdx.x < st) {
            redA[threadIdx.x] += redA[threadIdx.x + st];
            redB[threadIdx.x] += redB[threadIdx.x + st];
        }
        __syncthreads();
    }
    float mu  = redA[0] / (float)L;
    float var = redB[0] / (float)L - mu * mu;
    float rs  = rsqrtf(var + 1e-5f);
    for (int i = threadIdx.x; i < L; i += 256) {
        float v = ((float)xr[i] - mu) * rs * g[i] + bb[i];
        out[row * (size_t)L + i] = (bf16)v;
    }
}

// ---------------------------------------------------------------------------
// RoPE (interleaved pairs) in-place on q,k halves of qkv (bf16)
// ---------------------------------------------------------------------------
__global__ __launch_bounds__(256) void rope_kernel(bf16* __restrict__ qkv)
{
    const int N = 2048, C = 1024, C3 = 3072;
    int gid   = blockIdx.x * blockDim.x + threadIdx.x; // 4194304 total
    int i     = gid & 31;
    int h     = (gid >> 5) & 15;
    int which = (gid >> 9) & 1;
    int token = gid >> 10;
    int pos   = token & (N - 1);
    // 10000^(-2i/64) = 2^(-i * log2(10000)/32)
    float inv = exp2f(-(float)i * 0.41524101186092029f);
    float ang = (float)pos * inv;
    float c = cosf(ang), sn = sinf(ang);
    size_t base = (size_t)token * C3 + (size_t)which * C + h * 64 + 2 * i;
    float x0 = (float)qkv[base], x1 = (float)qkv[base + 1];
    qkv[base]     = (bf16)(x0 * c - x1 * sn);
    qkv[base + 1] = (bf16)(x1 * c + x0 * sn);
}

// ---------------------------------------------------------------------------
// Flash-style attention, double-buffered K/V stream.
// WG = (128-query block, b*H+h), 256 thr = 8 waves; wave owns 16 query rows.
// 64-key tiles, 16 wmma / iter, two barriers / iter.
// ---------------------------------------------------------------------------
__global__ __launch_bounds__(256) void attn_kernel(
    const bf16* __restrict__ qkv, bf16* __restrict__ o)
{
    constexpr int Dh = 64, Bq = 128, Bkk = 64, QP = 72, KP = 72, VP = 72, PP = 72;
    const int N = 2048, C = 1024, C3 = 3072, H = 16;
    __shared__ bf16 Qs[Bq * QP];
    __shared__ bf16 Ks[2][Bkk * KP];
    __shared__ bf16 Vts[2][Dh * VP];   // transposed: [d][key]
    __shared__ bf16 Ps[8][16 * PP];    // per-wave P tile (16 x 64)

    const int tid = threadIdx.x;
    const int l   = tid & 31;
    const int w   = tid >> 5;
    const int qb  = blockIdx.x;
    const int bh  = blockIdx.y;
    const int b   = bh / H, h = bh % H;
    const size_t tok0 = (size_t)b * N + qb * Bq;

    const int fr = l & 15;
    const int kc = (l >> 4) << 3;
    const int kb = (l >> 4) << 4;
    const int rh = (l >> 4) << 3;

    auto load_kv = [&](int kt, int buf) {
        size_t ktok = (size_t)b * N + (size_t)kt * Bkk;
        // K tile 64x64 (async)
#pragma unroll
        for (int p = 0; p < 2; p++) {
            int c = p * 256 + tid;
            int row = c >> 3, c8 = (c & 7) * 8;
            copy16(&qkv[(ktok + row) * C3 + C + h * Dh + c8], &Ks[buf][row * KP + c8]);
        }
        // V tile 64x64 -> Vts[d][key] (transpose on the way in)
#pragma unroll
        for (int p = 0; p < 4; p++) {
            int lin = p * 256 + tid;
            int key = lin >> 4, c4 = (lin & 15) * 4;
            const bf16* src = &qkv[(ktok + key) * C3 + 2 * C + h * Dh + c4];
            Vts[buf][(c4 + 0) * VP + key] = src[0];
            Vts[buf][(c4 + 1) * VP + key] = src[1];
            Vts[buf][(c4 + 2) * VP + key] = src[2];
            Vts[buf][(c4 + 3) * VP + key] = src[3];
        }
    };

    // Q tile 128x64
#pragma unroll
    for (int p = 0; p < 4; p++) {
        int c = p * 256 + tid;
        int row = c >> 3, c8 = (c & 7) * 8;
        copy16(&qkv[(tok0 + row) * C3 + h * Dh + c8], &Qs[row * QP + c8]);
    }
    load_kv(0, 0);
    async_fence();
    __syncthreads();

    Frag aQ[2];
#pragma unroll
    for (int s = 0; s < 2; s++) {
        const bf16* base = &Qs[(w * 16 + fr) * QP + s * 32 + kc];
        aQ[s].u[0] = *(const uint4*)base;
        aQ[s].u[1] = *(const uint4*)(base + 16);
    }

    float mrow[8], lrow[8];
    v8f oacc[4];
#pragma unroll
    for (int r = 0; r < 8; r++) { mrow[r] = -1e30f; lrow[r] = 0.f; }
#pragma unroll
    for (int jd = 0; jd < 4; jd++) oacc[jd] = {};
    const float scale = 0.125f;

    const int nk = N / Bkk;
    for (int kt = 0; kt < nk; kt++) {
        if (kt) {
            async_fence();
            __syncthreads();                 // tile kt published; buf kt-1 free
        }
        if (kt + 1 < nk) load_kv(kt + 1, (kt + 1) & 1);
        const int buf = kt & 1;

        // S = Q K^T : 4 key tiles x 2 k-steps
        v8f s[4];
#pragma unroll
        for (int j = 0; j < 4; j++) s[j] = {};
#pragma unroll
        for (int j = 0; j < 4; j++)
#pragma unroll
            for (int ks = 0; ks < 2; ks++) {
                Frag bK;
                const bf16* base = &Ks[buf][(j * 16 + fr) * KP + ks * 32 + kc];
                bK.u[0] = *(const uint4*)base;
                bK.u[1] = *(const uint4*)(base + 16);
                s[j] = __builtin_amdgcn_wmma_f32_16x16x32_bf16(
                    false, aQ[ks].v, false, bK.v, (short)0, s[j], false, false);
            }

        // online softmax (per query row; 16-lane butterflies)
        float alpha[8];
#pragma unroll
        for (int r = 0; r < 8; r++) {
            float v0 = fmaxf(fmaxf(s[0][r], s[1][r]), fmaxf(s[2][r], s[3][r])) * scale;
            for (int m = 1; m < 16; m <<= 1) v0 = fmaxf(v0, __shfl_xor(v0, m, 32));
            float mnew = fmaxf(mrow[r], v0);
            alpha[r] = __expf(mrow[r] - mnew);
            float p0 = __expf(s[0][r] * scale - mnew);
            float p1 = __expf(s[1][r] * scale - mnew);
            float p2 = __expf(s[2][r] * scale - mnew);
            float p3 = __expf(s[3][r] * scale - mnew);
            float psum = (p0 + p1) + (p2 + p3);
            for (int m = 1; m < 16; m <<= 1) psum += __shfl_xor(psum, m, 32);
            lrow[r] = lrow[r] * alpha[r] + psum;
            mrow[r] = mnew;
            bf16* pr = &Ps[w][(rh + r) * PP];
            pr[fr]      = (bf16)p0;
            pr[16 + fr] = (bf16)p1;
            pr[32 + fr] = (bf16)p2;
            pr[48 + fr] = (bf16)p3;
        }

        // P A-fragments (16x64 -> 2 k-steps)
        Frag aP[2];
#pragma unroll
        for (int ks = 0; ks < 2; ks++) {
            const bf16* pb = &Ps[w][fr * PP + ks * 32 + kc];
            aP[ks].u[0] = *(const uint4*)pb;
            aP[ks].u[1] = *(const uint4*)(pb + 16);
        }

        // O = alpha*O + P V
#pragma unroll
        for (int jd = 0; jd < 4; jd++) {
            v8f t = oacc[jd];
#pragma unroll
            for (int r = 0; r < 8; r++) t[r] *= alpha[r];
#pragma unroll
            for (int ks = 0; ks < 2; ks++) {
                Frag bV;
                const bf16* vb = &Vts[buf][(jd * 16 + fr) * VP + ks * 32 + kc];
                bV.u[0] = *(const uint4*)vb;
                bV.u[1] = *(const uint4*)(vb + 16);
                t = __builtin_amdgcn_wmma_f32_16x16x32_bf16(
                    false, aP[ks].v, false, bV.v, (short)0, t, false, false);
            }
            oacc[jd] = t;
        }
        __syncthreads();                     // waves done with buf kt before overwrite
    }

    // normalize + store
#pragma unroll
    for (int jd = 0; jd < 4; jd++)
#pragma unroll
        for (int r = 0; r < 8; r++) {
            int row = w * 16 + rh + r;
            int d   = jd * 16 + fr;
            float val = oacc[jd][r] / lrow[r];
            o[(tok0 + row) * C + h * Dh + d] = (bf16)val;
        }
}

// ---------------------------------------------------------------------------
// Host launcher
// ---------------------------------------------------------------------------
extern "C" void kernel_launch(void* const* d_in, const int* in_sizes, int n_in,
                              void* d_out, int out_size, void* d_ws, size_t ws_size,
                              hipStream_t stream)
{
    (void)in_sizes; (void)n_in; (void)out_size; (void)ws_size;
    const int Bv = 2, Nv = 2048, Cv = 1024, HIDv = 4096;
    const int M = Bv * Nv;

    const float* x      = (const float*)d_in[0];
    const float* ln1_g  = (const float*)d_in[1];
    const float* ln1_b  = (const float*)d_in[2];
    const float* qkv_w  = (const float*)d_in[3];
    const float* qkv_b  = (const float*)d_in[4];
    const float* proj_w = (const float*)d_in[5];
    const float* proj_b = (const float*)d_in[6];
    const float* ln2_g  = (const float*)d_in[7];
    const float* ln2_b  = (const float*)d_in[8];
    const float* fc1_w  = (const float*)d_in[9];
    const float* fc1_b  = (const float*)d_in[10];
    const float* ffng   = (const float*)d_in[11];
    const float* ffnb   = (const float*)d_in[12];
    const float* fc2_w  = (const float*)d_in[13];
    const float* fc2_b  = (const float*)d_in[14];
    float* out = (float*)d_out;

    const size_t MB = 1024 * 1024;
    char* ws = (char*)d_ws;
    bf16*  hx      = (bf16*)(ws + 0 * MB);     //  8 MB [4096,1024]
    bf16*  qkv     = (bf16*)(ws + 8 * MB);     // 24 MB [4096,3072]
    bf16*  obuf    = (bf16*)(ws + 32 * MB);    //  8 MB [4096,1024]
    float* x1      = (float*)(ws + 40 * MB);   // 16 MB [4096,1024]
    bf16*  mbuf    = (bf16*)(ws + 56 * MB);    //  8 MB [4096,1024]
    bf16*  hbuf    = (bf16*)(ws + 64 * MB);    // 32 MB [4096,4096]
    bf16*  h2      = (bf16*)(ws + 96 * MB);    // 32 MB [4096,4096]
    bf16*  wt_qkv  = (bf16*)(ws + 128 * MB);   //  6 MB [3072,1024]
    bf16*  wt_proj = (bf16*)(ws + 136 * MB);   //  2 MB [1024,1024]
    bf16*  wt_fc1  = (bf16*)(ws + 138 * MB);   //  8 MB [4096,1024]
    bf16*  wt_fc2  = (bf16*)(ws + 146 * MB);   //  8 MB [1024,4096]

    // 0) weight prep: f32[K,N] -> bf16[N,K]
    wprep_kernel<<<dim3(3 * Cv / 32, Cv / 32), 256, 0, stream>>>(qkv_w, wt_qkv, Cv, 3 * Cv);
    wprep_kernel<<<dim3(Cv / 32, Cv / 32), 256, 0, stream>>>(proj_w, wt_proj, Cv, Cv);
    wprep_kernel<<<dim3(HIDv / 32, Cv / 32), 256, 0, stream>>>(fc1_w, wt_fc1, Cv, HIDv);
    wprep_kernel<<<dim3(Cv / 32, HIDv / 32), 256, 0, stream>>>(fc2_w, wt_fc2, HIDv, Cv);

    // 1) LN1 -> bf16
    ln_kernel<float><<<M, 256, 0, stream>>>(x, ln1_g, ln1_b, hx, Cv);
    // 2) QKV GEMM (+bias) -> bf16
    gemm_bf16<0><<<dim3(3 * Cv / BN, M / BM), 256, 0, stream>>>(
        hx, wt_qkv, qkv_b, nullptr, qkv, M, 3 * Cv, Cv);
    // 3) RoPE in place on q,k
    rope_kernel<<<(M * 2 * 16 * 32) / 256, 256, 0, stream>>>(qkv);
    // 4) attention -> obuf
    attn_kernel<<<dim3(Nv / 128, Bv * 16), 256, 0, stream>>>(qkv, obuf);
    // 5) proj (+bias+residual x) -> x1 fp32
    gemm_bf16<1><<<dim3(Cv / BN, M / BM), 256, 0, stream>>>(
        obuf, wt_proj, proj_b, x, x1, M, Cv, Cv);
    // 6) LN2 -> bf16
    ln_kernel<float><<<M, 256, 0, stream>>>(x1, ln2_g, ln2_b, mbuf, Cv);
    // 7) FC1 (+bias, GELU) -> bf16
    gemm_bf16<2><<<dim3(HIDv / BN, M / BM), 256, 0, stream>>>(
        mbuf, wt_fc1, fc1_b, nullptr, hbuf, M, HIDv, Cv);
    // 8) FFN LN on hidden -> bf16
    ln_kernel<bf16><<<M, 256, 0, stream>>>(hbuf, ffng, ffnb, h2, HIDv);
    // 9) FC2 (+bias+residual x1) -> d_out fp32
    gemm_bf16<1><<<dim3(Cv / BN, M / BM), 256, 0, stream>>>(
        h2, wt_fc2, fc2_b, x1, out, M, Cv, HIDv);
}
